// CompressAttn_88235808129266
// MI455X (gfx1250) — matrix operand order repaired and checked by reading, not compiled, not run on hardware
//
#include <hip/hip_runtime.h>
#include <hip/hip_bf16.h>

typedef _Float16 h8  __attribute__((ext_vector_type(8)));
typedef _Float16 h16 __attribute__((ext_vector_type(16)));
typedef float    f8  __attribute__((ext_vector_type(8)));
typedef int      i32x4 __attribute__((vector_size(16)));

#define B_    4
#define S_    4096
#define HQ_   16
#define DQK_  128
#define DV_   128
#define KER_  32
#define STR_  16
#define TOPK_ 16
#define C_    255
#define CP_   256
#define SM_SCALE 0.08838834764831845f   // 128^-0.5

#ifndef __has_builtin
#define __has_builtin(x) 0
#endif
#if __has_builtin(__builtin_amdgcn_global_load_async_to_lds_b128)
#define HAS_ASYNC_LDS 1
#else
#define HAS_ASYNC_LDS 0
#endif

// 16-byte copy global -> LDS. Async (ASYNCcnt-tracked, no VGPR bounce) when the
// gfx1250 async builtin is available, otherwise plain load+store.
static __device__ __forceinline__ void cp16(const _Float16* g, _Float16* l) {
#if HAS_ASYNC_LDS
  __builtin_amdgcn_global_load_async_to_lds_b128(
      (__attribute__((address_space(1))) i32x4*)(uintptr_t)g,
      (__attribute__((address_space(3))) i32x4*)(uint32_t)(uintptr_t)l,
      0, 0);
#else
  *(h8*)l = *(const h8*)g;
#endif
}

static __device__ __forceinline__ void async_wait_all() {
#if HAS_ASYNC_LDS
#if __has_builtin(__builtin_amdgcn_s_wait_asynccnt)
  __builtin_amdgcn_s_wait_asynccnt(0);
#else
  asm volatile("s_wait_asynccnt 0x0" ::: "memory");
#endif
#endif
}

// ---------------------------------------------------------------------------
// Kernel 1: gate-compress K and V windows.
//   ckh : [B][CP][128] f16  (row-major over c -> WMMA A tiles for S = CK x Q^T)
//   cvth: [B][128][CP] f16  (d-major        -> WMMA A tiles for O^T = CV^T x P^T)
// Rows/cols with c >= C are zero-filled padding.
// ---------------------------------------------------------------------------
__global__ __launch_bounds__(128)
void nsa_compress(const float* __restrict__ kbuf, const float* __restrict__ vbuf,
                  const float* __restrict__ gwk, const float* __restrict__ gwv,
                  const int* __restrict__ r2t,
                  _Float16* __restrict__ ckh, _Float16* __restrict__ cvth) {
  const int b = blockIdx.x >> 8;
  const int c = blockIdx.x & 255;
  const int t = threadIdx.x;
  if (c >= C_) {
    ckh[((size_t)(b * CP_ + c)) * DQK_ + t] = (_Float16)0.f;
    cvth[((size_t)(b * DQK_ + t)) * CP_ + c] = (_Float16)0.f;
    return;
  }
  __shared__ float sK[KER_][DQK_];
  __shared__ float sV[KER_][DQK_];
  __shared__ float sPart[2][KER_][4];
  __shared__ float sG[2][KER_];

  { // stage the 32x128 K and V windows
    const int kr = t >> 2;
    const int d0 = (t & 3) * 32;
    const int tok = r2t[b * S_ + c * STR_ + kr];
    const float4* ksrc = (const float4*)(kbuf + (size_t)tok * DQK_ + d0);
    const float4* vsrc = (const float4*)(vbuf + (size_t)tok * DQK_ + d0);
    float4* kdst = (float4*)&sK[kr][d0];
    float4* vdst = (float4*)&sV[kr][d0];
#pragma unroll
    for (int j = 0; j < 8; ++j) { kdst[j] = ksrc[j]; vdst[j] = vsrc[j]; }
  }
  __syncthreads();
  { // gate logits: L[kk] = sum_{k,d} win[k][d] * W[kk][k*128+d], 4 partials each
    const int kk = t & 31;
    const int part = t >> 5;
    float aK = 0.f, aV = 0.f;
    for (int kr = part * 8; kr < part * 8 + 8; ++kr) {
      const float4* gk = (const float4*)(gwk + (size_t)kk * (KER_ * DQK_) + kr * DQK_);
      const float4* gv = (const float4*)(gwv + (size_t)kk * (KER_ * DQK_) + kr * DQK_);
      const float4* xk = (const float4*)&sK[kr][0];
      const float4* xv = (const float4*)&sV[kr][0];
#pragma unroll 4
      for (int j = 0; j < 32; ++j) {
        float4 a = gk[j], x = xk[j];
        aK += a.x * x.x + a.y * x.y + a.z * x.z + a.w * x.w;
        float4 av = gv[j], xv4 = xv[j];
        aV += av.x * xv4.x + av.y * xv4.y + av.z * xv4.z + av.w * xv4.w;
      }
    }
    sPart[0][kk][part] = aK;
    sPart[1][kk][part] = aV;
  }
  __syncthreads();
  if (t < 64) { // softmax over 32 gate logits (wave0: K, wave1: V)
    const int sel = t >> 5;
    const int kk = t & 31;
    float v = sPart[sel][kk][0] + sPart[sel][kk][1] + sPart[sel][kk][2] + sPart[sel][kk][3];
    float mx = v;
#pragma unroll
    for (int off = 16; off; off >>= 1) mx = fmaxf(mx, __shfl_xor(mx, off));
    float e = __expf(v - mx);
    float sum = e;
#pragma unroll
    for (int off = 16; off; off >>= 1) sum += __shfl_xor(sum, off);
    sG[sel][kk] = e / sum;
  }
  __syncthreads();
  { // weighted sums, emit f16 CK row and CV^T column
    const int d = t;
    float ck = 0.f, cv = 0.f;
#pragma unroll 8
    for (int kr = 0; kr < KER_; ++kr) {
      ck += sG[0][kr] * sK[kr][d];
      cv += sG[1][kr] * sV[kr][d];
    }
    ckh[((size_t)(b * CP_ + c)) * DQK_ + d] = (_Float16)ck;
    cvth[((size_t)(b * DQK_ + d)) * CP_ + c] = (_Float16)cv;
  }
}

// Cooperative chunk staging: 32 CK rows (and optionally 32 CV^T columns of all
// 128 d-rows). 256 threads x 32B each buffer. Async when available.
static __device__ __forceinline__ void stage_chunk(int tid,
    const _Float16* __restrict__ ckg, const _Float16* __restrict__ cvg,
    _Float16 (*ck)[136], _Float16 (*cvt)[40], int c0, bool withV) {
  const int row = tid >> 3;
  const int col = (tid & 7) << 4;
  const _Float16* src = ckg + (size_t)(c0 + row) * DQK_ + col;
  cp16(src, &ck[row][col]);
  cp16(src + 8, &ck[row][col + 8]);
  if (withV) {
    const int dr = tid >> 1;
    const int cc = (tid & 1) << 4;
    const _Float16* vsrc = cvg + (size_t)dr * CP_ + c0 + cc;
    cp16(vsrc, &cvt[dr][cc]);
    cp16(vsrc + 8, &cvt[dr][cc + 8]);
  }
}

// ---------------------------------------------------------------------------
// Kernel 2: WMMA attention + block-score selection.
// One wave per query token (16 heads = WMMA N dim). 8 waves/block share
// double-buffered, async-staged CK / CV^T chunks of 32 compressed blocks.
// ---------------------------------------------------------------------------
__global__ __launch_bounds__(256)
void nsa_attn(const float* __restrict__ q,
              const _Float16* __restrict__ ckh,
              const _Float16* __restrict__ cvth,
              float* __restrict__ out_o,
              float* __restrict__ out_idx) {
  __shared__ _Float16 sCK[2][KER_][136];   // ping-pong 32 x (128+8 pad)
  __shared__ _Float16 sCVT[2][DQK_][40];   // ping-pong 128 x (32+8 pad)
  __shared__ _Float16 sP[8][16][40];       // per wave: P as [head][c_local]
  __shared__ float    sScore[8][CP_];      // per wave: block scores

  const int tid  = threadIdx.x;
  const int wave = tid >> 5;
  const int lane = tid & 31;
  const int hidx = lane & 15;      // head (WMMA column)
  const int hi   = lane >> 4;      // half-wave id
  const int kb   = hi * 8;         // A-layout K base offset

  const int gid0 = blockIdx.x * 8;
  const int gid  = gid0 + wave;
  const int b    = gid >> 12;
  const int s    = gid & (S_ - 1);
  const int s0   = gid0 & (S_ - 1);

  const int cmax   = (s >= 31) ? ((s - 31) >> 4) : -1;          // last visible block
  const int cmaxg  = ((s0 + 7) >= 31) ? ((s0 + 7 - 31) >> 4) : -1;
  const int nchunk = (cmaxg + 32) >> 5;

  // Q^T in WMMA B layout, loop-invariant: lane holds head hidx, k = kc*32 + hi*8 + e (e<8), kc*32 + 16 + hi*8 + (e-8) (e>=8)
  h16 qb[4];
  {
    const float* qp = q + (size_t)gid * (HQ_ * DQK_) + hidx * DQK_ + hi * 8;
#pragma unroll
    for (int kc = 0; kc < 4; ++kc) {
      const float4* src = (const float4*)(qp + kc * 32);
#pragma unroll
      for (int j = 0; j < 4; ++j) {
        float4 v = src[j + (j >> 1) * 2];
        qb[kc][j * 4 + 0] = (_Float16)v.x;
        qb[kc][j * 4 + 1] = (_Float16)v.y;
        qb[kc][j * 4 + 2] = (_Float16)v.z;
        qb[kc][j * 4 + 3] = (_Float16)v.w;
      }
    }
  }

  const _Float16* ckg = ckh + (size_t)b * CP_ * DQK_;
  const _Float16* cvg = cvth + (size_t)b * DQK_ * CP_;

  // ---------------- pass 1: row max + sum (online) ----------------
  float mrun = -__builtin_inff();
  float lrun = 0.f;
  if (nchunk > 0) stage_chunk(tid, ckg, cvg, sCK[0], sCVT[0], 0, false);
  for (int ch = 0; ch < nchunk; ++ch) {
    const int c0 = ch << 5;
    const int cur = ch & 1;
    async_wait_all();
    __syncthreads();
    if (ch + 1 < nchunk)
      stage_chunk(tid, ckg, cvg, sCK[cur ^ 1], sCVT[cur ^ 1], (ch + 1) << 5, false);
    if (c0 > cmax) continue;   // wave-uniform: fully past causal horizon
#pragma unroll
    for (int hf = 0; hf < 2; ++hf) {
      const int ct0 = c0 + hf * 16;
      if (ct0 > cmax) continue;
      h16 a4[4];
#pragma unroll
      for (int kc = 0; kc < 4; ++kc) {
        const int off = kc * 32 + kb;
        h8 lo = *(const h8*)&sCK[cur][hf * 16 + hidx][off];
        h8 up = *(const h8*)&sCK[cur][hf * 16 + hidx][off + 16];
        a4[kc] = __builtin_shufflevector(lo, up, 0,1,2,3,4,5,6,7,8,9,10,11,12,13,14,15);
      }
      f8 acc = {0.f, 0.f, 0.f, 0.f, 0.f, 0.f, 0.f, 0.f};
#pragma unroll
      for (int kc = 0; kc < 4; ++kc)
        acc = __builtin_amdgcn_wmma_f32_16x16x32_f16(false, a4[kc], false, qb[kc],
                                                     (short)0, acc, false, false);
      float sv[8];
      float tmax = -__builtin_inff();
#pragma unroll
      for (int r = 0; r < 8; ++r) {
        int c = ct0 + r + kb;
        sv[r] = (c <= cmax) ? acc[r] * SM_SCALE : -__builtin_inff();
        tmax = fmaxf(tmax, sv[r]);
      }
      tmax = fmaxf(tmax, __shfl_xor(tmax, 16));
      float mnew = fmaxf(mrun, tmax);
      float corr = __expf(mrun - mnew);
      float tsum = 0.f;
#pragma unroll
      for (int r = 0; r < 8; ++r) tsum += __expf(sv[r] - mnew);
      tsum += __shfl_xor(tsum, 16);
      lrun = lrun * corr + tsum;
      mrun = mnew;
    }
  }
  const float linv = (lrun > 0.f) ? 1.f / lrun : 0.f;
  const float mfin = mrun;

  f8 oacc[8];
#pragma unroll
  for (int dt = 0; dt < 8; ++dt) oacc[dt] = (f8){0.f, 0.f, 0.f, 0.f, 0.f, 0.f, 0.f, 0.f};
#pragma unroll
  for (int j = 0; j < 8; ++j) sScore[wave][j * 32 + lane] = 0.f;

  // ---------------- pass 2: normalized P, PV GEMM, block scores ----------------
  __syncthreads();   // pass-1 reads done before re-staging buffer 0
  if (nchunk > 0) stage_chunk(tid, ckg, cvg, sCK[0], sCVT[0], 0, true);
  for (int ch = 0; ch < nchunk; ++ch) {
    const int c0 = ch << 5;
    const int cur = ch & 1;
    async_wait_all();
    __syncthreads();
    if (ch + 1 < nchunk)
      stage_chunk(tid, ckg, cvg, sCK[cur ^ 1], sCVT[cur ^ 1], (ch + 1) << 5, true);
    if (c0 > cmax) continue;   // wave-uniform skip: nothing visible, scores stay 0
#pragma unroll
    for (int hf = 0; hf < 2; ++hf) {
      const int ct0 = c0 + hf * 16;
      if (ct0 <= cmax) {
        h16 a4[4];
#pragma unroll
        for (int kc = 0; kc < 4; ++kc) {
          const int off = kc * 32 + kb;
          h8 lo = *(const h8*)&sCK[cur][hf * 16 + hidx][off];
          h8 up = *(const h8*)&sCK[cur][hf * 16 + hidx][off + 16];
          a4[kc] = __builtin_shufflevector(lo, up, 0,1,2,3,4,5,6,7,8,9,10,11,12,13,14,15);
        }
        f8 acc = {0.f, 0.f, 0.f, 0.f, 0.f, 0.f, 0.f, 0.f};
#pragma unroll
        for (int kc = 0; kc < 4; ++kc)
          acc = __builtin_amdgcn_wmma_f32_16x16x32_f16(false, a4[kc], false, qb[kc],
                                                       (short)0, acc, false, false);
        h8 pv;
#pragma unroll
        for (int r = 0; r < 8; ++r) {
          int c = ct0 + r + kb;
          float sl = (c <= cmax) ? acc[r] * SM_SCALE : -__builtin_inff();
          pv[r] = (_Float16)(__expf(sl - mfin) * linv);
        }
        *(h8*)&sP[wave][hidx][hf * 16 + kb] = pv;   // one b128 DS store
      } else {
        *(h8*)&sP[wave][hidx][hf * 16 + kb] =
            (h8){(_Float16)0.f, (_Float16)0.f, (_Float16)0.f, (_Float16)0.f,
                 (_Float16)0.f, (_Float16)0.f, (_Float16)0.f, (_Float16)0.f};
      }
    }
    // PV: O^T[d,h] += CV^T[d, c-chunk] x P^T[c-chunk, h]
    {
      h8 plo = *(const h8*)&sP[wave][hidx][kb];
      h8 phi = *(const h8*)&sP[wave][hidx][kb + 16];
      h16 pb = __builtin_shufflevector(plo, phi, 0,1,2,3,4,5,6,7,8,9,10,11,12,13,14,15);
#pragma unroll
      for (int dt = 0; dt < 8; ++dt) {
        const int dr = dt * 16 + hidx;
        h8 lo = *(const h8*)&sCVT[cur][dr][kb];
        h8 up = *(const h8*)&sCVT[cur][dr][kb + 16];
        h16 a = __builtin_shufflevector(lo, up, 0,1,2,3,4,5,6,7,8,9,10,11,12,13,14,15);
        oacc[dt] = __builtin_amdgcn_wmma_f32_16x16x32_f16(false, a, false, pb,
                                                          (short)0, oacc[dt], false, false);
      }
    }
    { // block score: sum over 16 heads of p (lane = c_local)
      float ssum = 0.f;
#pragma unroll
      for (int hh = 0; hh < 16; ++hh) ssum += (float)sP[wave][hh][lane];
      sScore[wave][c0 + lane] = ssum;
    }
  }

  // ---------------- store O (lane: head hidx, 8 consecutive d per tile half) ----------------
  {
    const size_t obase = (size_t)gid * (HQ_ * DV_) + (size_t)hidx * DV_;
#pragma unroll
    for (int dt = 0; dt < 8; ++dt) {
      float4 v0, v1;
      v0.x = oacc[dt][0]; v0.y = oacc[dt][1]; v0.z = oacc[dt][2]; v0.w = oacc[dt][3];
      v1.x = oacc[dt][4]; v1.y = oacc[dt][5]; v1.z = oacc[dt][6]; v1.w = oacc[dt][7];
      float* dst = out_o + obase + dt * 16 + kb;
      *(float4*)dst = v0;
      *(float4*)(dst + 4) = v1;
    }
  }

  // ---------------- selection: slc[64] + top-16 (tie -> lowest index) ----------------
  const int cur_m = s >> 6;
  float vsl[2];
  int msl[2] = {lane, lane + 32};
#pragma unroll
  for (int tI = 0; tI < 2; ++tI) {
    int m = msl[tI];
    int cb = m << 2;
    float v = sScore[wave][cb] + sScore[wave][cb + 1] + sScore[wave][cb + 2];
    if (m >= 1) v += 0.5f * sScore[wave][cb - 1];
    if (cb + 3 < C_) v += 0.5f * sScore[wave][cb + 3];
    if (m > cur_m) v = -__FLT_MAX__;
    else if (m < 1 || m >= cur_m - 1) v = __FLT_MAX__;
    vsl[tI] = v;
  }
  float* idxout = out_idx + (size_t)gid * TOPK_;
  for (int i = 0; i < TOPK_; ++i) {
    float bv; int bi;
    if (vsl[0] > vsl[1] || (vsl[0] == vsl[1] && msl[0] < msl[1])) { bv = vsl[0]; bi = msl[0]; }
    else { bv = vsl[1]; bi = msl[1]; }
#pragma unroll
    for (int off = 1; off < 32; off <<= 1) {
      float ov = __shfl_xor(bv, off);
      int   oi = __shfl_xor(bi, off);
      if (ov > bv || (ov == bv && oi < bi)) { bv = ov; bi = oi; }
    }
    if (lane == 0) idxout[i] = (float)bi;
    if (bi == msl[0]) vsl[0] = -__builtin_inff();
    if (bi == msl[1]) vsl[1] = -__builtin_inff();
  }
}

extern "C" void kernel_launch(void* const* d_in, const int* in_sizes, int n_in,
                              void* d_out, int out_size, void* d_ws, size_t ws_size,
                              hipStream_t stream) {
  const float* q    = (const float*)d_in[0];
  const float* kbuf = (const float*)d_in[1];
  const float* vbuf = (const float*)d_in[2];
  const float* gwk  = (const float*)d_in[3];
  const float* gwv  = (const float*)d_in[4];
  const int*   r2t  = (const int*)d_in[5];

  _Float16* ckh  = (_Float16*)d_ws;                       // [B][CP][128] f16
  _Float16* cvth = ckh + (size_t)B_ * CP_ * DQK_;         // [B][128][CP] f16
  float* out_o   = (float*)d_out;                         // [B*S, HQ*DV]
  float* out_idx = out_o + (size_t)B_ * S_ * HQ_ * DV_;   // [B*S, 1, 16] as float

  nsa_compress<<<dim3(B_ * CP_), dim3(128), 0, stream>>>(kbuf, vbuf, gwk, gwv, r2t, ckh, cvth);
  nsa_attn<<<dim3((B_ * S_) / 8), dim3(256), 0, stream>>>(q, ckh, cvth, out_o, out_idx);
}
